// GCN_52450140619484
// MI455X (gfx1250) — matrix operand order repaired and checked by reading, not compiled, not run on hardware
//
#include <hip/hip_runtime.h>
#include <math.h>

typedef __attribute__((ext_vector_type(16))) __bf16 v16bf;
typedef __attribute__((ext_vector_type(8)))  float  v8f;

#define N_FEAT1 1024
#define D_HID   128
#define D_OUT   9

// ---------------------------------------------------------------------------
// Degree / normalization
// ---------------------------------------------------------------------------
__global__ void deg_init_kernel(float* __restrict__ deg, int n) {
    int i = blockIdx.x * blockDim.x + threadIdx.x;
    if (i < n) deg[i] = 1.0f;  // self loop contributes 1
}

__global__ void deg_edge_kernel(const int* __restrict__ dst, float* __restrict__ deg, int e) {
    int i = blockIdx.x * blockDim.x + threadIdx.x;
    if (i < e) atomicAdd(&deg[dst[i]], 1.0f);
}

__global__ void dinv_kernel(float* __restrict__ deg, int n) {
    int i = blockIdx.x * blockDim.x + threadIdx.x;
    if (i < n) deg[i] = rsqrtf(deg[i]);   // deg > 0 always (self loops)
}

// ---------------------------------------------------------------------------
// Weight pre-pass: W[K x C] fp32 row-major -> Wt[Cpad x K] bf16 (col-major,
// zero-padded cols). Makes each lane's WMMA B-fragment 16 *contiguous* bf16
// = one 32-byte load instead of 16 strided b32 loads.
// ---------------------------------------------------------------------------
__global__ void wt_convert_kernel(const float* __restrict__ W, __bf16* __restrict__ Wt,
                                  int K, int C, int Cpad) {
    int idx = blockIdx.x * blockDim.x + threadIdx.x;
    if (idx >= Cpad * K) return;
    int col = idx / K;
    int k   = idx - col * K;
    Wt[idx] = (col < C) ? (__bf16)W[(size_t)k * C + col] : (__bf16)0.0f;
}

// ---------------------------------------------------------------------------
// WMMA GEMM, C[N x 128] = A[N x K] * Bt^T, Bt is bf16 [128 x K] col-major.
// Block = 256 threads = 8 waves; wave w owns column tile w (16 cols).
// Block covers 16 rows; grid.x = N/16 (N is a multiple of 16).
// ---------------------------------------------------------------------------
__global__ void gemm128_wmma_kernel(const float* __restrict__ A,
                                    const __bf16* __restrict__ Bt,
                                    float* __restrict__ C, int K) {
    const int lane = threadIdx.x & 31;
    const int wave = threadIdx.x >> 5;           // 0..7 -> column tile
    const int half = lane >> 4;                  // 0 or 1
    const int row  = lane & 15;
    const int col  = wave * 16 + (lane & 15);    // global output column
    const float*  arow = A + (size_t)(blockIdx.x * 16 + row) * K;
    const __bf16* bcol = Bt + (size_t)col * K + half * 16;

    v8f acc = {};
    for (int k0 = 0; k0 < K; k0 += 32) {
        // A fragment: two contiguous runs of 8 floats, cvt to bf16
        const float4* p1 = (const float4*)(arow + k0 + half * 16);
        const float4* p2 = (const float4*)(arow + k0 + half * 16 + 8);
        float4 f0 = p1[0], f1 = p1[1], f2 = p2[0], f3 = p2[1];
        if (k0 + 32 < K) __builtin_prefetch(arow + k0 + 64, 0, 3);
        v16bf a;
        a[0]=(__bf16)f0.x; a[1]=(__bf16)f0.y; a[2]=(__bf16)f0.z; a[3]=(__bf16)f0.w;
        a[4]=(__bf16)f1.x; a[5]=(__bf16)f1.y; a[6]=(__bf16)f1.z; a[7]=(__bf16)f1.w;
        a[8]=(__bf16)f2.x; a[9]=(__bf16)f2.y; a[10]=(__bf16)f2.z; a[11]=(__bf16)f2.w;
        a[12]=(__bf16)f3.x; a[13]=(__bf16)f3.y; a[14]=(__bf16)f3.z; a[15]=(__bf16)f3.w;
        // B fragment: 16 contiguous bf16 at this lane's column, K = k0+16h..+15
        v16bf b = *(const v16bf*)(bcol + k0);
        acc = __builtin_amdgcn_wmma_f32_16x16x32_bf16(
            false, a, false, b, (short)0, acc, false, false);
    }
    const int rbase = blockIdx.x * 16 + half * 8;
    #pragma unroll
    for (int r = 0; r < 8; ++r)
        C[(size_t)(rbase + r) * D_HID + col] = acc[r];
}

// ---------------------------------------------------------------------------
// Final-layer WMMA GEMM: C[N x 9] = A[N x 128] * W3. Bt is bf16 [16 x 128]
// col-major, zero-padded cols 9..15; masked store of cols < 9.
// Block = 256 threads = 8 waves; wave w -> row tile blockIdx.x*8 + w.
// ---------------------------------------------------------------------------
__global__ void gemm9_wmma_kernel(const float* __restrict__ A,
                                  const __bf16* __restrict__ Bt,
                                  float* __restrict__ C, int nRowTiles) {
    const int lane = threadIdx.x & 31;
    const int wave = threadIdx.x >> 5;
    const int rowTile = blockIdx.x * 8 + wave;
    if (rowTile >= nRowTiles) return;            // uniform per wave -> EXEC stays all-1
    const int half = lane >> 4;
    const int row  = lane & 15;
    const int col  = lane & 15;
    const float*  arow = A + (size_t)(rowTile * 16 + row) * D_HID;
    const __bf16* bcol = Bt + (size_t)col * D_HID + half * 16;

    v8f acc = {};
    #pragma unroll
    for (int k0 = 0; k0 < D_HID; k0 += 32) {
        const float4* p1 = (const float4*)(arow + k0 + half * 16);
        const float4* p2 = (const float4*)(arow + k0 + half * 16 + 8);
        float4 f0 = p1[0], f1 = p1[1], f2 = p2[0], f3 = p2[1];
        v16bf a;
        a[0]=(__bf16)f0.x; a[1]=(__bf16)f0.y; a[2]=(__bf16)f0.z; a[3]=(__bf16)f0.w;
        a[4]=(__bf16)f1.x; a[5]=(__bf16)f1.y; a[6]=(__bf16)f1.z; a[7]=(__bf16)f1.w;
        a[8]=(__bf16)f2.x; a[9]=(__bf16)f2.y; a[10]=(__bf16)f2.z; a[11]=(__bf16)f2.w;
        a[12]=(__bf16)f3.x; a[13]=(__bf16)f3.y; a[14]=(__bf16)f3.z; a[15]=(__bf16)f3.w;
        v16bf b = *(const v16bf*)(bcol + k0);
        acc = __builtin_amdgcn_wmma_f32_16x16x32_bf16(
            false, a, false, b, (short)0, acc, false, false);
    }
    if (col < D_OUT) {
        const int rbase = rowTile * 16 + half * 8;
        #pragma unroll
        for (int r = 0; r < 8; ++r)
            C[(size_t)(rbase + r) * D_OUT + col] = acc[r];
    }
}

// ---------------------------------------------------------------------------
// Aggregation: self-loop init, edge scatter-add, bias(+ReLU)
// ---------------------------------------------------------------------------
__global__ void selfloop_init_kernel(const float* __restrict__ h,
                                     const float* __restrict__ dinv,
                                     float* __restrict__ out, int n, int d) {
    int idx = blockIdx.x * blockDim.x + threadIdx.x;
    if (idx >= n * d) return;
    int node = idx / d;
    float w = dinv[node];
    out[idx] = h[idx] * w * w;
}

__global__ void scatter_edge_kernel(const int* __restrict__ src,
                                    const int* __restrict__ dst,
                                    const float* __restrict__ dinv,
                                    const float* __restrict__ h,
                                    float* __restrict__ out, int e, int d) {
    long long idx = (long long)blockIdx.x * blockDim.x + threadIdx.x;
    if (idx >= (long long)e * d) return;
    int edge = (int)(idx / d);
    int f    = (int)(idx - (long long)edge * d);
    int s = src[edge], t = dst[edge];
    float norm = dinv[s] * dinv[t];
    atomicAdd(&out[(size_t)t * d + f], h[(size_t)s * d + f] * norm);
}

__global__ void bias_act_kernel(float* __restrict__ h, const float* __restrict__ bias,
                                int total, int d, int do_relu) {
    int idx = blockIdx.x * blockDim.x + threadIdx.x;
    if (idx >= total) return;
    float v = h[idx] + bias[idx % d];
    h[idx] = do_relu ? fmaxf(v, 0.0f) : v;
}

// ---------------------------------------------------------------------------
// Final: add bias, log-softmax over the 9 classes, write to d_out.
// ---------------------------------------------------------------------------
__global__ void bias_logsoftmax_kernel(const float* __restrict__ g,
                                       const float* __restrict__ bias,
                                       float* __restrict__ out, int n) {
    int i = blockIdx.x * blockDim.x + threadIdx.x;
    if (i >= n) return;
    float v[D_OUT];
    float m = -INFINITY;
    #pragma unroll
    for (int f = 0; f < D_OUT; ++f) {
        v[f] = g[(size_t)i * D_OUT + f] + bias[f];
        m = fmaxf(m, v[f]);
    }
    float s = 0.0f;
    #pragma unroll
    for (int f = 0; f < D_OUT; ++f) s += __expf(v[f] - m);
    float ls = __logf(s);
    #pragma unroll
    for (int f = 0; f < D_OUT; ++f)
        out[(size_t)i * D_OUT + f] = v[f] - m - ls;
}

// ---------------------------------------------------------------------------
extern "C" void kernel_launch(void* const* d_in, const int* in_sizes, int n_in,
                              void* d_out, int out_size, void* d_ws, size_t ws_size,
                              hipStream_t stream) {
    const float* x  = (const float*)d_in[0];
    const int*   ei = (const int*)d_in[1];
    const float* W1 = (const float*)d_in[2];
    const float* b1 = (const float*)d_in[3];
    const float* W2 = (const float*)d_in[4];
    const float* b2 = (const float*)d_in[5];
    const float* W3 = (const float*)d_in[6];
    const float* b3 = (const float*)d_in[7];

    const int N = in_sizes[0] / N_FEAT1;   // 100000
    const int E = in_sizes[1] / 2;         // 3200000
    const int* src = ei;
    const int* dst = ei + E;

    float* ws   = (float*)d_ws;
    float* dinv = ws;                       // [N]
    float* hA   = dinv + N;                 // [N * 128]
    float* hB   = hA + (size_t)N * D_HID;   // [N * 128]
    float* h9   = hA;                       // alias: hA free after layer-2 agg
    float* g9   = hA + (size_t)N * 16;      // alias, disjoint from h9
    __bf16* Wt1 = (__bf16*)(hB + (size_t)N * D_HID);  // [128 * 1024]
    __bf16* Wt2 = Wt1 + (size_t)D_HID * N_FEAT1;      // [128 * 128]
    __bf16* Wt3 = Wt2 + (size_t)D_HID * D_HID;        // [16 * 128], zero-padded

    const int T = 256;
    const int nTilesRow = N / 16;           // 100000 % 16 == 0

    // weight pre-conversion (tiny; L2-resident thereafter)
    wt_convert_kernel<<<(D_HID * N_FEAT1 + T - 1) / T, T, 0, stream>>>(W1, Wt1, N_FEAT1, D_HID, D_HID);
    wt_convert_kernel<<<(D_HID * D_HID + T - 1) / T, T, 0, stream>>>(W2, Wt2, D_HID, D_HID, D_HID);
    wt_convert_kernel<<<(16 * D_HID + T - 1) / T, T, 0, stream>>>(W3, Wt3, D_HID, D_OUT, 16);

    // normalization
    deg_init_kernel<<<(N + T - 1) / T, T, 0, stream>>>(dinv, N);
    deg_edge_kernel<<<(E + T - 1) / T, T, 0, stream>>>(dst, dinv, E);
    dinv_kernel<<<(N + T - 1) / T, T, 0, stream>>>(dinv, N);

    long long ed = (long long)E * D_HID;
    int edBlocks = (int)((ed + T - 1) / T);

    // ---- layer 1: X[N,1024] @ W1 -> hA; aggregate -> hB; +b1, ReLU
    gemm128_wmma_kernel<<<nTilesRow, T, 0, stream>>>(x, Wt1, hA, N_FEAT1);
    selfloop_init_kernel<<<(N * D_HID + T - 1) / T, T, 0, stream>>>(hA, dinv, hB, N, D_HID);
    scatter_edge_kernel<<<edBlocks, T, 0, stream>>>(src, dst, dinv, hA, hB, E, D_HID);
    bias_act_kernel<<<(N * D_HID + T - 1) / T, T, 0, stream>>>(hB, b1, N * D_HID, D_HID, 1);

    // ---- layer 2: hB @ W2 -> hA; aggregate -> hB; +b2, ReLU
    gemm128_wmma_kernel<<<nTilesRow, T, 0, stream>>>(hB, Wt2, hA, D_HID);
    selfloop_init_kernel<<<(N * D_HID + T - 1) / T, T, 0, stream>>>(hA, dinv, hB, N, D_HID);
    scatter_edge_kernel<<<edBlocks, T, 0, stream>>>(src, dst, dinv, hA, hB, E, D_HID);
    bias_act_kernel<<<(N * D_HID + T - 1) / T, T, 0, stream>>>(hB, b2, N * D_HID, D_HID, 1);

    // ---- layer 3: hB @ W3 -> h9; aggregate -> g9; +b3, log_softmax -> d_out
    gemm9_wmma_kernel<<<(nTilesRow + 7) / 8, T, 0, stream>>>(hB, Wt3, h9, nTilesRow);
    selfloop_init_kernel<<<(N * D_OUT + T - 1) / T, T, 0, stream>>>(h9, dinv, g9, N, D_OUT);
    long long ed9 = (long long)E * D_OUT;
    scatter_edge_kernel<<<(int)((ed9 + T - 1) / T), T, 0, stream>>>(src, dst, dinv, h9, g9, E, D_OUT);
    bias_logsoftmax_kernel<<<(N + T - 1) / T, T, 0, stream>>>(g9, b3, (float*)d_out, N);
}
